// GPT2Block_50766513439228
// MI455X (gfx1250) — matrix operand order repaired and checked by reading, not compiled, not run on hardware
//
#include <hip/hip_runtime.h>
#include <hip/hip_bf16.h>

// ---------------- constants ----------------
#define BATCH 2
#define SEQ   2048
#define DMODEL 1024
#define NHEAD 16
#define HDIM  64
#define DFF   4096
#define LN_EPS 1e-5f

typedef __attribute__((ext_vector_type(16))) __bf16 v16bf;
typedef __attribute__((ext_vector_type(8)))  float  v8f;
typedef __attribute__((ext_vector_type(4)))  unsigned int u32x4;
typedef __attribute__((ext_vector_type(8)))  unsigned int u32x8;

union BF16x16 { v16bf v; uint4 u[2]; };

__device__ __forceinline__ v8f wmma_bf16(v16bf a, v16bf b, v8f c) {
  return __builtin_amdgcn_wmma_f32_16x16x32_bf16(
      /*neg_a=*/false, a, /*neg_b=*/false, b,
      /*c_mod=*/(short)0, c, /*reuse_a=*/false, /*reuse_b=*/false);
}

// ---------------- weight convert + transpose (fp32 RxC -> bf16 CxR) ----------------
__global__ void transpose_cvt(const float* __restrict__ src, __bf16* __restrict__ dst,
                              int R, int C) {
  __shared__ float tile[32][33];
  int rb = blockIdx.x * 32, cb = blockIdx.y * 32;
  int tx = threadIdx.x & 31, ty = threadIdx.x >> 5;   // 256 threads: ty = 0..7
#pragma unroll
  for (int i = ty; i < 32; i += 8)
    tile[i][tx] = src[(size_t)(rb + i) * C + cb + tx];
  __syncthreads();
#pragma unroll
  for (int i = ty; i < 32; i += 8)
    dst[(size_t)(cb + i) * R + rb + tx] = (__bf16)tile[tx][i];
}

// ---------------- LayerNorm fp32 -> bf16 ----------------
__global__ void ln_kernel(const float* __restrict__ x,
                          const float* __restrict__ scale,
                          const float* __restrict__ bias,
                          __bf16* __restrict__ out) {
  const int D = DMODEL;
  int row = blockIdx.x;
  const float* xr = x + (size_t)row * D;
  __shared__ float redS[8], redS2[8];
  float s = 0.f, s2 = 0.f;
  for (int i = threadIdx.x; i < D; i += blockDim.x) { float v = xr[i]; s += v; s2 += v * v; }
#pragma unroll
  for (int o = 16; o >= 1; o >>= 1) { s += __shfl_xor(s, o, 32); s2 += __shfl_xor(s2, o, 32); }
  int w = threadIdx.x >> 5;
  if ((threadIdx.x & 31) == 0) { redS[w] = s; redS2[w] = s2; }
  __syncthreads();
  s = 0.f; s2 = 0.f;
#pragma unroll
  for (int i = 0; i < 8; i++) { s += redS[i]; s2 += redS2[i]; }
  float mu = s / D;
  float var = s2 / D - mu * mu;
  float rstd = rsqrtf(var + LN_EPS);
  for (int i = threadIdx.x; i < D; i += blockDim.x)
    out[(size_t)row * D + i] = (__bf16)((xr[i] - mu) * rstd * scale[i] + bias[i]);
}

// ---------------- WMMA GEMM with TDM-staged A operand ----------------
// C(MxN) = A(MxK bf16) * BT(NxK bf16)^T + bias
// Block: 128 threads = 4 waves stacked along M. Wave tile: 32(M) x 64(N) = 8 accumulators.
// Each wave DMAs its own 32x32 A panel into LDS with tensor_load_to_lds (double buffered),
// synchronized with its private TENSORcnt.
#define EPI_QKV      0
#define EPI_ATTNPROJ 1
#define EPI_FC       2
#define EPI_MLP      3

template <int EPI>
__global__ void gemm_bf16_kernel(const __bf16* __restrict__ A,
                                 const __bf16* __restrict__ BT,
                                 const float* __restrict__ bias,
                                 int M, int N, int K,
                                 const float* __restrict__ res,
                                 float* __restrict__ outF,
                                 __bf16* __restrict__ outB,
                                 __bf16* __restrict__ outQ,
                                 __bf16* __restrict__ outK,
                                 __bf16* __restrict__ outV) {
  // per-wave double-buffered A panel: 32 rows x 32 k, row stride 64B
  __shared__ __align__(16) __bf16 Alds[4][2][32][32];   // 16 KB

  int wave  = threadIdx.x >> 5;
  int lane  = threadIdx.x & 31;
  int lrow  = lane & 15;
  int lhalf = lane >> 4;
  int m0 = blockIdx.x * 128 + wave * 32;
  int n0 = blockIdx.y * 64;

  // ---- TDM descriptor group 1 (constant across K loop) ----
  // data_size=2B (code 1); tensor_dim0=K, tensor_dim1=M; tile 32(k) x 32(rows);
  // tensor_dim0_stride=K elements.
  u32x8 g1;
  g1[0] = 1u << 16;                                    // data_size=1 (2 bytes)
  g1[1] = ((unsigned)K & 0xFFFFu) << 16;               // tensor_dim0[15:0]
  g1[2] = (((unsigned)K >> 16) & 0xFFFFu) | (((unsigned)M & 0xFFFFu) << 16);
  g1[3] = (((unsigned)M >> 16) & 0xFFFFu) | (32u << 16);  // tile_dim0 = 32
  g1[4] = 32u;                                         // tile_dim1 = 32 rows
  g1[5] = (unsigned)K;                                 // tensor_dim0_stride lo
  g1[6] = 0u;
  g1[7] = 0u;

  auto tdm_issue = [&](int k0, int buf) {
    unsigned long long ga =
        (unsigned long long)(uintptr_t)(A + (size_t)m0 * K + k0);
    unsigned lds = (unsigned)(uintptr_t)&Alds[wave][buf][0][0];
    u32x4 g0;
    g0[0] = 1u;                                        // count=1, user descriptor
    g0[1] = lds;                                       // lds_addr
    g0[2] = (unsigned)ga;                              // global_addr[31:0]
    g0[3] = (unsigned)(ga >> 32) | (2u << 30);         // global_addr[56:32] | type=2
    asm volatile("tensor_load_to_lds %0, %1" :: "s"(g0), "s"(g1) : "memory");
  };

  v8f acc[2][4] = {};
  tdm_issue(0, 0);
  for (int k0 = 0; k0 < K; k0 += 32) {
    int buf = (k0 >> 5) & 1;
    if (k0 + 32 < K) {
      tdm_issue(k0 + 32, buf ^ 1);
      __builtin_amdgcn_s_wait_tensorcnt(1);            // current buffer ready
    } else {
      __builtin_amdgcn_s_wait_tensorcnt(0);
    }

    BF16x16 a[2], b[4];
#pragma unroll
    for (int i = 0; i < 2; i++) {                       // A from LDS
      const __bf16* ap = &Alds[wave][buf][16 * i + lrow][lhalf * 8];
      a[i].u[0] = *(const uint4*)ap;
      a[i].u[1] = *(const uint4*)(ap + 16);
    }
#pragma unroll
    for (int j = 0; j < 4; j++) {                       // B from global (L2-resident)
      const __bf16* bp = BT + (size_t)(n0 + 16 * j + lrow) * K + k0 + lhalf * 8;
      b[j].u[0] = *(const uint4*)bp;
      b[j].u[1] = *(const uint4*)(bp + 16);
      __builtin_prefetch(bp + 32, 0, 1);
    }
#pragma unroll
    for (int i = 0; i < 2; i++)
#pragma unroll
      for (int j = 0; j < 4; j++)
        acc[i][j] = wmma_bf16(a[i].v, b[j].v, acc[i][j]);
  }

#pragma unroll
  for (int i = 0; i < 2; i++) {
#pragma unroll
    for (int j = 0; j < 4; j++) {
#pragma unroll
      for (int r = 0; r < 8; r++) {
        int m = m0 + 16 * i + r + 8 * lhalf;
        int n = n0 + 16 * j + lrow;
        float val = acc[i][j][r] + bias[n];
        if constexpr (EPI == EPI_QKV) {
          int sect = n >> 10;           // 0=q 1=k 2=v
          int d = n & 1023;
          int h = d >> 6, hd = d & 63;
          int bb = m >> 11, s = m & 2047;
          size_t bh = (size_t)bb * NHEAD + h;
          if (sect == 0)      outQ[(bh * SEQ + s) * HDIM + hd] = (__bf16)(val * 0.125f);
          else if (sect == 1) outK[(bh * SEQ + s) * HDIM + hd] = (__bf16)val;
          else                outV[(bh * HDIM + hd) * SEQ + s] = (__bf16)val;  // V transposed
        } else if constexpr (EPI == EPI_ATTNPROJ) {
          size_t idx = (size_t)m * N + n;
          outF[idx] = val + res[idx];           // x2 = x + attn_proj
        } else if constexpr (EPI == EPI_FC) {
          float g = 0.5f * val * (1.0f + tanhf(0.7978845608028654f *
                                               (val + 0.044715f * val * val * val)));
          outB[(size_t)m * N + n] = (__bf16)g;
        } else {  // EPI_MLP
          size_t idx = (size_t)m * N + n;
          outF[idx] = val + res[idx];           // out = x2 + mlp_proj
        }
      }
    }
  }
}

// ---------------- causal flash attention, one wave per 16 queries ----------------
__global__ void attn_kernel(const __bf16* __restrict__ q,
                            const __bf16* __restrict__ k,
                            const __bf16* __restrict__ vT,
                            __bf16* __restrict__ attn) {
  int bh = blockIdx.x;                 // 0..B*H-1
  int q0 = blockIdx.y * 16;
  int b  = bh / NHEAD, h = bh % NHEAD;
  int lane = threadIdx.x & 31, lrow = lane & 15, lhalf = lane >> 4;

  const __bf16* qp = q  + (size_t)bh * SEQ * HDIM;
  const __bf16* kp = k  + (size_t)bh * SEQ * HDIM;
  const __bf16* vp = vT + (size_t)bh * HDIM * SEQ;

  // Q as two 16x32 A-tiles (head-dim split 0..31 / 32..63); Q pre-scaled by 1/sqrt(64)
  BF16x16 qa[2];
#pragma unroll
  for (int t = 0; t < 2; t++) {
    const __bf16* ap = qp + (size_t)(q0 + lrow) * HDIM + 32 * t + lhalf * 8;
    qa[t].u[0] = *(const uint4*)ap;
    qa[t].u[1] = *(const uint4*)(ap + 16);
  }

  float mrow[8], srow[8];
#pragma unroll
  for (int r = 0; r < 8; r++) { mrow[r] = -3.0e38f; srow[r] = 0.f; }
  int qend = q0 + 16;

  // ---- pass 1: row max + row sum (online) ----
  for (int j0 = 0; j0 < qend; j0 += 16) {
    v8f sc = {};
#pragma unroll
    for (int t = 0; t < 2; t++) {
      BF16x16 kb;
      const __bf16* bp = kp + (size_t)(j0 + lrow) * HDIM + 32 * t + lhalf * 8;
      kb.u[0] = *(const uint4*)bp;
      kb.u[1] = *(const uint4*)(bp + 16);
      sc = wmma_bf16(qa[t].v, kb.v, sc);
    }
    int kj = j0 + lrow;
#pragma unroll
    for (int r = 0; r < 8; r++) {
      int qi = q0 + r + 8 * lhalf;
      float v = (kj <= qi) ? sc[r] : -1e30f;
      float tm = v;
#pragma unroll
      for (int o = 8; o >= 1; o >>= 1) tm = fmaxf(tm, __shfl_xor(tm, o, 16));
      float mnew = fmaxf(mrow[r], tm);
      float e = __expf(v - mnew);
#pragma unroll
      for (int o = 8; o >= 1; o >>= 1) e += __shfl_xor(e, o, 16);
      srow[r] = srow[r] * __expf(mrow[r] - mnew) + e;
      mrow[r] = mnew;
    }
  }

  // ---- pass 2: recompute P, transpose via LDS, accumulate P*V ----
  __shared__ __align__(16) __bf16 Plds[16][40];
  v8f oacc[4] = {};
  for (int j0 = 0; j0 < qend; j0 += 32) {
#pragma unroll
    for (int t = 0; t < 2; t++) {
      v8f sc = {};
#pragma unroll
      for (int u = 0; u < 2; u++) {
        BF16x16 kb;
        const __bf16* bp = kp + (size_t)(j0 + 16 * t + lrow) * HDIM + 32 * u + lhalf * 8;
        kb.u[0] = *(const uint4*)bp;
        kb.u[1] = *(const uint4*)(bp + 16);
        sc = wmma_bf16(qa[u].v, kb.v, sc);
      }
      int kj = j0 + 16 * t + lrow;
#pragma unroll
      for (int r = 0; r < 8; r++) {
        int qi = q0 + r + 8 * lhalf;
        float v = (kj <= qi) ? sc[r] : -1e30f;
        Plds[r + 8 * lhalf][16 * t + lrow] = (__bf16)__expf(v - mrow[r]);
      }
    }
    __syncthreads();
    BF16x16 pa;
    const __bf16* pp = &Plds[lrow][lhalf * 8];
    pa.u[0] = *(const uint4*)pp;
    pa.u[1] = *(const uint4*)(pp + 16);
#pragma unroll
    for (int n = 0; n < 4; n++) {
      BF16x16 vb;
      const __bf16* vpp = vp + (size_t)(16 * n + lrow) * SEQ + j0 + lhalf * 8;
      vb.u[0] = *(const uint4*)vpp;
      vb.u[1] = *(const uint4*)(vpp + 16);
      oacc[n] = wmma_bf16(pa.v, vb.v, oacc[n]);
    }
    __syncthreads();
  }

  // normalize + store to (b, s, d) bf16
#pragma unroll
  for (int n = 0; n < 4; n++) {
#pragma unroll
    for (int r = 0; r < 8; r++) {
      float o = oacc[n][r] / srow[r];
      int srowi = q0 + r + 8 * lhalf;
      int d = h * HDIM + 16 * n + lrow;
      attn[((size_t)b * SEQ + srowi) * DMODEL + d] = (__bf16)o;
    }
  }
}

// ---------------- launch ----------------
extern "C" void kernel_launch(void* const* d_in, const int* in_sizes, int n_in,
                              void* d_out, int out_size, void* d_ws, size_t ws_size,
                              hipStream_t stream) {
  const float* x        = (const float*)d_in[0];
  // d_in[1] = mask (implicit causal, unused)
  const float* ln1s     = (const float*)d_in[2];
  const float* ln1b     = (const float*)d_in[3];
  const float* ln2s     = (const float*)d_in[4];
  const float* ln2b     = (const float*)d_in[5];
  const float* w_qkv    = (const float*)d_in[6];
  const float* b_qkv    = (const float*)d_in[7];
  const float* w_attn   = (const float*)d_in[8];
  const float* b_attn   = (const float*)d_in[9];
  const float* w_fc     = (const float*)d_in[10];
  const float* b_fc     = (const float*)d_in[11];
  const float* w_mlp    = (const float*)d_in[12];
  const float* b_mlp    = (const float*)d_in[13];
  float* out = (float*)d_out;

  const int M = BATCH * SEQ;             // 4096
  char* ws = (char*)d_ws;
  size_t off = 0;
  auto alloc = [&](size_t bytes) { void* p = ws + off; off = (off + bytes + 255) & ~(size_t)255; return p; };

  __bf16* wqkvT = (__bf16*)alloc((size_t)3 * DMODEL * DMODEL * 2);
  __bf16* wattT = (__bf16*)alloc((size_t)DMODEL * DMODEL * 2);
  __bf16* wfcT  = (__bf16*)alloc((size_t)DFF * DMODEL * 2);
  __bf16* wmlpT = (__bf16*)alloc((size_t)DMODEL * DFF * 2);
  __bf16* h1    = (__bf16*)alloc((size_t)M * DMODEL * 2);
  __bf16* qb    = (__bf16*)alloc((size_t)M * DMODEL * 2);
  __bf16* kb    = (__bf16*)alloc((size_t)M * DMODEL * 2);
  __bf16* vTb   = (__bf16*)alloc((size_t)M * DMODEL * 2);
  __bf16* attnb = (__bf16*)alloc((size_t)M * DMODEL * 2);
  float*  x2    = (float*) alloc((size_t)M * DMODEL * 4);
  __bf16* h2    = (__bf16*)alloc((size_t)M * DMODEL * 2);
  __bf16* hfc   = (__bf16*)alloc((size_t)M * DFF * 2);

  // 1) weights -> bf16 transposed
  transpose_cvt<<<dim3(DMODEL / 32, 3 * DMODEL / 32), 256, 0, stream>>>(w_qkv, wqkvT, DMODEL, 3 * DMODEL);
  transpose_cvt<<<dim3(DMODEL / 32, DMODEL / 32),     256, 0, stream>>>(w_attn, wattT, DMODEL, DMODEL);
  transpose_cvt<<<dim3(DMODEL / 32, DFF / 32),        256, 0, stream>>>(w_fc, wfcT, DMODEL, DFF);
  transpose_cvt<<<dim3(DFF / 32, DMODEL / 32),        256, 0, stream>>>(w_mlp, wmlpT, DFF, DMODEL);

  // 2) LN1
  ln_kernel<<<M, 256, 0, stream>>>(x, ln1s, ln1b, h1);

  // 3) QKV GEMM (M=4096, N=3072, K=1024) with q-scale / v-transpose epilogue
  gemm_bf16_kernel<EPI_QKV><<<dim3(M / 128, 3 * DMODEL / 64), 128, 0, stream>>>(
      h1, wqkvT, b_qkv, M, 3 * DMODEL, DMODEL, nullptr, nullptr, nullptr, qb, kb, vTb);

  // 4) causal attention
  attn_kernel<<<dim3(BATCH * NHEAD, SEQ / 16), 32, 0, stream>>>(qb, kb, vTb, attnb);

  // 5) attn proj + residual -> x2 (fp32)
  gemm_bf16_kernel<EPI_ATTNPROJ><<<dim3(M / 128, DMODEL / 64), 128, 0, stream>>>(
      attnb, wattT, b_attn, M, DMODEL, DMODEL, x, x2, nullptr, nullptr, nullptr, nullptr);

  // 6) LN2
  ln_kernel<<<M, 256, 0, stream>>>(x2, ln2s, ln2b, h2);

  // 7) FC + GELU (M=4096, N=4096, K=1024)
  gemm_bf16_kernel<EPI_FC><<<dim3(M / 128, DFF / 64), 128, 0, stream>>>(
      h2, wfcT, b_fc, M, DFF, DMODEL, nullptr, nullptr, hfc, nullptr, nullptr, nullptr);

  // 8) MLP proj + residual -> out (M=4096, N=1024, K=4096)
  gemm_bf16_kernel<EPI_MLP><<<dim3(M / 128, DMODEL / 64), 128, 0, stream>>>(
      hfc, wmlpT, b_mlp, M, DMODEL, DFF, x2, out, nullptr, nullptr, nullptr, nullptr);

  (void)in_sizes; (void)n_in; (void)out_size; (void)ws_size;
}
